// BahdanauAttention_34711925686481
// MI455X (gfx1250) — matrix operand order repaired and checked by reading, not compiled, not run on hardware
//
#include <hip/hip_runtime.h>
#include <hip/hip_bf16.h>
#include <math.h>

#define B_ 64
#define S_ 2048
#define H_ 1024
#define K_ 1024

typedef __attribute__((ext_vector_type(16))) __bf16 v16bf;
typedef __attribute__((ext_vector_type(8)))  float  v8f;
typedef __attribute__((ext_vector_type(8)))  __bf16 bf16x8;
typedef __attribute__((ext_vector_type(4)))  __bf16 bf16x4;

// Hardware tanh (V_TANH_F32 is a gfx1250 TRANS op) with safe fallbacks.
__device__ __forceinline__ float fast_tanhf(float x) {
#if __has_builtin(__builtin_amdgcn_tanhf)
    return __builtin_amdgcn_tanhf(x);
#elif __has_builtin(__builtin_amdgcn_tanh_f32)
    return __builtin_amdgcn_tanh_f32(x);
#else
    return tanhf(x);
#endif
}

// ---------------- kernel 0a: convert Wk (f32 [H,K]) -> bf16 ----------------
__global__ void wk_to_bf16_kernel(const float* __restrict__ wk,
                                  __bf16* __restrict__ wkbf) {
    int i = (blockIdx.x * blockDim.x + threadIdx.x) * 4;
    float4 f = *(const float4*)(wk + i);
    bf16x4 o;
    o[0] = (__bf16)f.x; o[1] = (__bf16)f.y; o[2] = (__bf16)f.z; o[3] = (__bf16)f.w;
    *(bf16x4*)(wkbf + i) = o;
}

// ---------------- kernel 0b: qproj[b,h] = query[b,:] . Wq[h,:] -------------
__global__ void qproj_kernel(const float* __restrict__ query,
                             const float* __restrict__ Wq,
                             float* __restrict__ qp) {
    __shared__ float sq[1024];
    const int b = blockIdx.x;
    for (int i = threadIdx.x; i < 1024; i += 256) sq[i] = query[(size_t)b * 1024 + i];
    __syncthreads();
    for (int h = threadIdx.x; h < 1024; h += 256) {
        const float* w = Wq + (size_t)h * 1024;
        float acc = 0.f;
        #pragma unroll 4
        for (int k = 0; k < 1024; k += 4) {
            float4 wv = *(const float4*)(w + k);
            acc += sq[k] * wv.x + sq[k + 1] * wv.y + sq[k + 2] * wv.z + sq[k + 3] * wv.w;
        }
        qp[(size_t)b * 1024 + h] = acc;
    }
}

// ---------------- kernel 0c: zero context region of d_out ------------------
__global__ void zero_ctx_kernel(float* __restrict__ p) {
    p[blockIdx.x * blockDim.x + threadIdx.x] = 0.f;
}

// ---------------- main fused kernel: energy[b,s] ---------------------------
// grid (S/64, B), block 256 (8 waves). Wave w: m-tile = w&3, h-parity = w>>2.
#define ROWPAD 1032   // 1024 + 8 bf16 pad: lane row-stride = 516 dwords -> 2-way max on ds_b128
__launch_bounds__(256, 1)
__global__ void energy_kernel(const float* __restrict__ keys,
                              const unsigned char* __restrict__ mask,
                              const __bf16* __restrict__ wkbf,
                              const float* __restrict__ qp,
                              const float* __restrict__ vvec,
                              float* __restrict__ energy) {
    __shared__ __bf16 sKeys[64 * ROWPAD];   // 129 KB: 64 s-rows x 1024 k (bf16)
    __shared__ float  sQp[1024];
    __shared__ float  sV[1024];
    __shared__ float  sEpart[2 * 64];       // per-h-parity partial energies

    const int b   = blockIdx.y;
    const int s0  = blockIdx.x * 64;
    const int tid = threadIdx.x;

    // Stage keys[b, s0:s0+64, :] f32 -> bf16 into LDS (each thread: 64 float4s)
    const float* kbase = keys + ((size_t)b * S_ + s0) * K_;
    for (int i = tid; i < 64 * 256; i += 256) {
        const int row = i >> 8;
        const int col = (i & 255) << 2;
        float4 f = *(const float4*)(kbase + (size_t)row * K_ + col);
        bf16x4 o;
        o[0] = (__bf16)f.x; o[1] = (__bf16)f.y; o[2] = (__bf16)f.z; o[3] = (__bf16)f.w;
        *(bf16x4*)(&sKeys[row * ROWPAD + col]) = o;
    }
    for (int i = tid; i < 1024; i += 256) {
        sQp[i] = qp[(size_t)b * 1024 + i];
        sV[i]  = vvec[i];
    }
    __syncthreads();

    const int  lane = tid & 31;
    const int  wave = tid >> 5;
    const int  m0   = (wave & 3) * 16;   // s-tile base within block
    const int  npar = wave >> 2;         // h-tile parity
    const int  lh   = lane & 15;
    const bool hiHalf = lane >= 16;

    // A fragment addressing (ISA 7.12.2, 16-bit A 16x32):
    // lanes 0-15: row m0+lh, K chunks {0..7},{16..23}; lanes 16-31: {8..15},{24..31}
    const int abase = (m0 + lh) * ROWPAD + (hiHalf ? 8 : 0);

    float eacc[8];
    #pragma unroll
    for (int r = 0; r < 8; ++r) eacc[r] = 0.f;

    for (int nt = npar; nt < 64; nt += 2) {
        const int h0 = nt * 16;
        // B fragment: lane reads two 8-bf16 chunks of Wk row (h0 + lh):
        // lanes 0-15 -> K {k0..k0+7},{k0+16..k0+23}; lanes 16-31 -> {k0+8..k0+15},{k0+24..k0+31}
        const __bf16* bbase = wkbf + (size_t)(h0 + lh) * K_ + (hiHalf ? 8 : 0);

        v8f c = {};
        #pragma unroll 8
        for (int kt = 0; kt < 32; ++kt) {
            const int ka = kt * 32;
            bf16x8 alo = *(const bf16x8*)(&sKeys[abase + ka]);
            bf16x8 ahi = *(const bf16x8*)(&sKeys[abase + ka + 16]);
            v16bf a = __builtin_shufflevector(alo, ahi,
                0, 1, 2, 3, 4, 5, 6, 7, 8, 9, 10, 11, 12, 13, 14, 15);
            bf16x8 blo = *(const bf16x8*)(bbase + ka);
            bf16x8 bhi = *(const bf16x8*)(bbase + ka + 16);
            v16bf bb = __builtin_shufflevector(blo, bhi,
                0, 1, 2, 3, 4, 5, 6, 7, 8, 9, 10, 11, 12, 13, 14, 15);
            c = __builtin_amdgcn_wmma_f32_16x16x32_bf16(
                    false, a, false, bb, (short)0, c, false, false);
        }

        // Epilogue: energy partial += v[h] * tanh(qp[h] + proj).
        // C layout: VGPR r -> row m0 + r + 8*hiHalf, col h0 + lh.
        const int   hc = h0 + lh;
        const float vh = sV[hc];
        const float qh = sQp[hc];
        #pragma unroll
        for (int r = 0; r < 8; ++r)
            eacc[r] += vh * fast_tanhf(qh + c[r]);
    }

    // Reduce over the 16 lanes of each half (columns of the C tile)
    #pragma unroll
    for (int r = 0; r < 8; ++r) {
        float x = eacc[r];
        x += __shfl_xor(x, 1, 32);
        x += __shfl_xor(x, 2, 32);
        x += __shfl_xor(x, 4, 32);
        x += __shfl_xor(x, 8, 32);
        eacc[r] = x;
    }
    // Deterministic combine: each of the two h-parity waves writes its slot.
    if (lh == 0) {
        const int rb = m0 + (hiHalf ? 8 : 0);
        #pragma unroll
        for (int r = 0; r < 8; ++r) sEpart[npar * 64 + rb + r] = eacc[r];
    }
    __syncthreads();

    if (tid < 64) {
        const int s = s0 + tid;
        float e = sEpart[tid] + sEpart[64 + tid];
        if (mask[(size_t)b * S_ + s]) e = -__builtin_inff();
        energy[(size_t)b * S_ + s] = e;
    }
}

// ---------------- softmax over S, in place ---------------------------------
__global__ void softmax_kernel(float* __restrict__ attn) {
    __shared__ float red[256];
    const int b = blockIdx.x, tid = threadIdx.x;
    float* e = attn + (size_t)b * S_;

    float m = -__builtin_inff();
    for (int s = tid; s < S_; s += 256) m = fmaxf(m, e[s]);
    red[tid] = m; __syncthreads();
    for (int off = 128; off; off >>= 1) {
        if (tid < off) red[tid] = fmaxf(red[tid], red[tid + off]);
        __syncthreads();
    }
    m = red[0];
    __syncthreads();

    const bool allpad = !(m > -__builtin_inff());
    float sum = 0.f;
    for (int s = tid; s < S_; s += 256) {
        float x = allpad ? 0.f : __expf(e[s] - m);
        e[s] = x;
        sum += x;
    }
    red[tid] = sum; __syncthreads();
    for (int off = 128; off; off >>= 1) {
        if (tid < off) red[tid] += red[tid + off];
        __syncthreads();
    }
    sum = red[0];
    const float inv = (sum > 0.f) ? 1.f / sum : 0.f;
    for (int s = tid; s < S_; s += 256) e[s] *= inv;
}

// ---------------- context[b,k] = sum_s attn[b,s] * keys[b,s,k] -------------
// grid (16 s-chunks, B); partials accumulated with hardware f32 atomics.
__global__ void context_kernel(const float* __restrict__ keys,
                               const float* __restrict__ attn,
                               float* __restrict__ ctx) {
    const int b  = blockIdx.y;
    const int sc = blockIdx.x;
    const int k  = threadIdx.x * 4;
    const float* kb = keys + ((size_t)b * S_ + sc * 128) * K_;
    const float* ab = attn + (size_t)b * S_ + sc * 128;
    float4 acc = {0.f, 0.f, 0.f, 0.f};
    for (int s = 0; s < 128; ++s) {
        const float a  = ab[s];
        const float4 kv = *(const float4*)(kb + (size_t)s * K_ + k);
        acc.x += a * kv.x; acc.y += a * kv.y; acc.z += a * kv.z; acc.w += a * kv.w;
    }
    float* o = ctx + (size_t)b * K_ + k;
    unsafeAtomicAdd(o + 0, acc.x);
    unsafeAtomicAdd(o + 1, acc.y);
    unsafeAtomicAdd(o + 2, acc.z);
    unsafeAtomicAdd(o + 3, acc.w);
}

extern "C" void kernel_launch(void* const* d_in, const int* in_sizes, int n_in,
                              void* d_out, int out_size, void* d_ws, size_t ws_size,
                              hipStream_t stream) {
    const float*         query = (const float*)d_in[0];          // [B,Q]
    const float*         keys  = (const float*)d_in[1];          // [B,S,K]
    const unsigned char* mask  = (const unsigned char*)d_in[2];  // [B,S] bool
    const float*         Wq    = (const float*)d_in[3];          // [H,Q]
    const float*         Wk    = (const float*)d_in[4];          // [H,K]
    const float*         v     = (const float*)d_in[5];          // [1,H]

    float* out  = (float*)d_out;
    float* ctx  = out;                 // [B,K] = 65536 floats
    float* attn = out + B_ * K_;       // [B,S] = 131072 floats

    char*   ws    = (char*)d_ws;
    float*  qp    = (float*)ws;                           // 256 KB
    __bf16* wkbf  = (__bf16*)(ws + (size_t)B_ * H_ * 4);  // 2 MB

    wk_to_bf16_kernel<<<(H_ * K_) / (256 * 4), 256, 0, stream>>>(Wk, wkbf);
    qproj_kernel<<<B_, 256, 0, stream>>>(query, Wq, qp);
    zero_ctx_kernel<<<(B_ * K_) / 256, 256, 0, stream>>>(ctx);
    energy_kernel<<<dim3(S_ / 64, B_), 256, 0, stream>>>(keys, mask, wkbf, qp, v, attn);
    softmax_kernel<<<B_, 256, 0, stream>>>(attn);
    context_kernel<<<dim3(16, B_), 256, 0, stream>>>(keys, attn, ctx);
}
